// RopelessMLA_41042707480979
// MI455X (gfx1250) — matrix operand order repaired and checked by reading, not compiled, not run on hardware
//
#include <hip/hip_runtime.h>

// ---------------------------------------------------------------------------
// RopelessMLA forward for MI455X (gfx1250), bf16 WMMA everywhere.
// D=2048, H=16, DH=128, L=512, B=2, S=2048.
// ---------------------------------------------------------------------------

typedef __attribute__((ext_vector_type(16))) __bf16 v16bf;
typedef __attribute__((ext_vector_type(8)))  float  v8f;
typedef __attribute__((ext_vector_type(4)))  int    v4i;

union Frag32B { uint4 u[2]; v16bf v; };

#if defined(__HIP_DEVICE_COMPILE__) &&                                   \
    __has_builtin(__builtin_amdgcn_global_load_async_to_lds_b128) &&     \
    __has_builtin(__builtin_amdgcn_s_wait_asynccnt)
#define HAVE_ASYNC_LDS 1
typedef __attribute__((address_space(1))) v4i v4i_g;  // global
typedef __attribute__((address_space(3))) v4i v4i_l;  // LDS
#else
#define HAVE_ASYNC_LDS 0
#endif

__device__ __forceinline__ v8f wmma_bf16(v16bf a, v16bf b, v8f c) {
  // D = A(16x32) * B(32x16) + C ; f32 accumulate
  return __builtin_amdgcn_wmma_f32_16x16x32_bf16(false, a, false, b,
                                                 (short)0, c, false, false);
}

// A-matrix 16x32 bf16 fragment (row m = lane&15; VGPR j<4: K=8*half+2j,
// j>=4: K=16+8*half+2(j-4))  -> two contiguous 16B loads per lane.
__device__ __forceinline__ v16bf frag_a(const __bf16* base, int row_base,
                                        int ld, int k0, int lane) {
  int m = row_base + (lane & 15);
  const __bf16* p = base + (size_t)m * ld + k0 + ((lane >> 4) << 3);
  Frag32B f;
  f.u[0] = *(const uint4*)(p);
  f.u[1] = *(const uint4*)(p + 16);
  return f.v;
}

// B-matrix 32x16 bf16 fragment where memory holds B^T rows: B[k][n] = M[n][k].
// lane: n = lane&15, elems 0..7: K=8*half+s, elems 8..15: K=16+8*half+(s-8).
__device__ __forceinline__ v16bf frag_b(const __bf16* base, int n_base,
                                        int ld, int k0, int lane) {
  int n = n_base + (lane & 15);
  const __bf16* p = base + (size_t)n * ld + k0 + ((lane >> 4) << 3);
  Frag32B f;
  f.u[0] = *(const uint4*)(p);
  f.u[1] = *(const uint4*)(p + 16);
  return f.v;
}

// 16-byte global -> LDS copy; async (ASYNCcnt-tracked) when available.
__device__ __forceinline__ void async_cp16(const __bf16* g, __bf16* l) {
#if HAVE_ASYNC_LDS
  __builtin_amdgcn_global_load_async_to_lds_b128(
      (v4i_g*)(g), (v4i_l*)(l), 0, 0);
#else
  *(uint4*)l = *(const uint4*)g;
#endif
}

__device__ __forceinline__ void async_wait0() {
#if HAVE_ASYNC_LDS
  __builtin_amdgcn_s_wait_asynccnt(0);
#endif
  __syncthreads();
}

// ---------------------------------------------------------------------------
// float -> bf16 convert
// ---------------------------------------------------------------------------
__global__ void f2bf_kernel(const float* __restrict__ in,
                            __bf16* __restrict__ out, size_t n) {
  size_t i = (size_t)blockIdx.x * blockDim.x + threadIdx.x;
  if (i < n) out[i] = (__bf16)in[i];
}

// in [R][C] f32 -> out [C][R] bf16   (transpose + convert)
__global__ void trans_f2bf_kernel(const float* __restrict__ in,
                                  __bf16* __restrict__ out, int R, int C) {
  size_t i = (size_t)blockIdx.x * blockDim.x + threadIdx.x;
  size_t total = (size_t)R * C;
  if (i < total) {
    int c = (int)(i / R);
    int r = (int)(i % R);
    out[i] = (__bf16)in[(size_t)r * C + c];
  }
}

// ---------------------------------------------------------------------------
// Generic batched WMMA GEMM:  C[M,N] = A[M,K] @ B[N,K]^T
// One wave per 64x64 C tile (16 accumulators; A frags reused over 4 B frags).
// batch z -> (b = z/hdiv, h = z%hdiv), operand offsets b*s?b + h*s?h (elems).
// ---------------------------------------------------------------------------
__global__ __launch_bounds__(32)
void gemm_bf16_wmma(const __bf16* __restrict__ A, const __bf16* __restrict__ B,
                    void* __restrict__ Cv, int K, int lda, int ldb, int ldc,
                    long long sAb, long long sAh, long long sBb, long long sBh,
                    long long sCb, long long sCh, int hdiv, int out_bf16) {
  const int lane = threadIdx.x;
  const int n0 = blockIdx.x * 64;
  const int m0 = blockIdx.y * 64;
  const int z = blockIdx.z;
  const int h = z % hdiv, b = z / hdiv;

  const __bf16* Ab = A + (size_t)b * sAb + (size_t)h * sAh;
  const __bf16* Bb = B + (size_t)b * sBb + (size_t)h * sBh;

  v8f acc[4][4] = {};
  for (int k0 = 0; k0 < K; k0 += 32) {
    v16bf af[4], bf[4];
    #pragma unroll
    for (int mi = 0; mi < 4; mi++) af[mi] = frag_a(Ab, m0 + mi * 16, lda, k0, lane);
    #pragma unroll
    for (int ni = 0; ni < 4; ni++) bf[ni] = frag_b(Bb, n0 + ni * 16, ldb, k0, lane);
    #pragma unroll
    for (int mi = 0; mi < 4; mi++)
      #pragma unroll
      for (int ni = 0; ni < 4; ni++)
        acc[mi][ni] = wmma_bf16(af[mi], bf[ni], acc[mi][ni]);
  }

  const int half = lane >> 4, l16 = lane & 15;
  if (out_bf16) {
    __bf16* C = (__bf16*)Cv + (size_t)b * sCb + (size_t)h * sCh;
    #pragma unroll
    for (int mi = 0; mi < 4; mi++)
      #pragma unroll
      for (int ni = 0; ni < 4; ni++)
        #pragma unroll
        for (int i = 0; i < 8; i++) {
          int row = m0 + mi * 16 + i + 8 * half;
          int col = n0 + ni * 16 + l16;
          C[(size_t)row * ldc + col] = (__bf16)acc[mi][ni][i];
        }
  } else {
    float* C = (float*)Cv + (size_t)b * sCb + (size_t)h * sCh;
    #pragma unroll
    for (int mi = 0; mi < 4; mi++)
      #pragma unroll
      for (int ni = 0; ni < 4; ni++)
        #pragma unroll
        for (int i = 0; i < 8; i++) {
          int row = m0 + mi * 16 + i + 8 * half;
          int col = n0 + ni * 16 + l16;
          C[(size_t)row * ldc + col] = acc[mi][ni][i];
        }
  }
}

// ---------------------------------------------------------------------------
// Row LayerNorm over L=512. One wave per row; 16 elems/lane.
// Writes f32 (output tuple) + bf16 (for downstream WMMA).
// ---------------------------------------------------------------------------
__global__ __launch_bounds__(32)
void ln_rows(const float* __restrict__ Hin, const float* __restrict__ g,
             const float* __restrict__ bt, float* __restrict__ outF,
             __bf16* __restrict__ outB) {
  const int row = blockIdx.x;
  const int lane = threadIdx.x;
  const float* p = Hin + (size_t)row * 512 + lane * 16;

  float x[16];
  #pragma unroll
  for (int q = 0; q < 4; q++) {
    float4 t = ((const float4*)p)[q];
    x[q * 4 + 0] = t.x; x[q * 4 + 1] = t.y;
    x[q * 4 + 2] = t.z; x[q * 4 + 3] = t.w;
  }
  float s = 0.f;
  #pragma unroll
  for (int j = 0; j < 16; j++) s += x[j];
  #pragma unroll
  for (int w = 1; w < 32; w <<= 1) s += __shfl_xor(s, w, 32);
  const float mu = s * (1.0f / 512.0f);

  float v = 0.f;
  #pragma unroll
  for (int j = 0; j < 16; j++) { float d = x[j] - mu; v += d * d; }
  #pragma unroll
  for (int w = 1; w < 32; w <<= 1) v += __shfl_xor(v, w, 32);
  const float rstd = rsqrtf(v * (1.0f / 512.0f) + 1e-5f);

  #pragma unroll
  for (int j = 0; j < 16; j++) {
    int c = lane * 16 + j;
    float y = (x[j] - mu) * rstd * g[c] + bt[c];
    outF[(size_t)row * 512 + c] = y;
    outB[(size_t)row * 512 + c] = (__bf16)y;
  }
}

// ---------------------------------------------------------------------------
// Flash attention over the latent KV, with ASYNCcnt-pipelined K/V staging.
// Q = tmp[b,h] : [S, 512] bf16 (VGPR-resident fragments)
// K = c_kv[b]  : [S, 512] bf16 -> double-buffered LDS tiles [32,512]
// V^T = vT[b, h*128+dv][t]     -> double-buffered LDS tiles [128,32]
// One wave per (b, h, 16-query tile). Online softmax; P staged via LDS.
// ---------------------------------------------------------------------------
__global__ __launch_bounds__(32)
void flash_mla(const __bf16* __restrict__ tmp, const __bf16* __restrict__ ckv,
               const __bf16* __restrict__ vT, __bf16* __restrict__ ctx) {
  const int S = 2048, L = 512, H = 16, DH = 128, D = 2048;
  const int lane = threadIdx.x;
  const int l16 = lane & 15, half = lane >> 4;
  const int q0 = blockIdx.x * 16;
  const int h = blockIdx.y;
  const int b = blockIdx.z;

  const __bf16* Q  = tmp + ((size_t)(b * H + h)) * S * L;
  const __bf16* Kc = ckv + (size_t)b * S * L;
  const __bf16* Vt = vT + ((size_t)b * D + (size_t)h * DH) * S;

  __shared__ __bf16 Kbuf[2][32 * 512];   // 2 x 32 KB
  __shared__ __bf16 Vbuf[2][128 * 32];   // 2 x 8 KB
  __shared__ __bf16 pT[16 * 32];

  // Preload Q fragments (16 K-chunks of 32) into VGPRs.
  v16bf qf[16];
  #pragma unroll
  for (int kc = 0; kc < 16; kc++) qf[kc] = frag_a(Q, q0, L, kc * 32, lane);

  v8f acc[8] = {};
  float mrow[8], lrow[8];
  #pragma unroll
  for (int i = 0; i < 8; i++) { mrow[i] = -1e30f; lrow[i] = 0.f; }

  const float sm = 0.08838834764831845f; // 1/sqrt(128)

  // Stage one 32-key tile (K: 32x512, V: 128x32) into LDS buffer `buf`.
  auto stage = [&](int t0, int buf) {
    #pragma unroll
    for (int it = 0; it < 64; it++) {           // K tile: 2048 x 16B chunks
      int c = it * 32 + lane;
      int r = c >> 6, o = (c & 63) * 8;
      async_cp16(Kc + (size_t)(t0 + r) * L + o, &Kbuf[buf][r * 512 + o]);
    }
    #pragma unroll
    for (int it = 0; it < 16; it++) {           // V tile: 512 x 16B chunks
      int c = it * 32 + lane;
      int r = c >> 2, o = (c & 3) * 8;
      async_cp16(Vt + (size_t)r * S + t0 + o, &Vbuf[buf][r * 32 + o]);
    }
  };

  stage(0, 0);
  async_wait0();

  for (int t0 = 0; t0 <= q0 + 15; t0 += 32) {
    const int cur = (t0 >> 5) & 1;
    if (t0 + 32 <= q0 + 15) stage(t0 + 32, cur ^ 1);  // overlap with compute

    // ---- scores: two 16x16 tiles over keys [t0, t0+31] ----
    v8f s0 = {}, s1 = {};
    #pragma unroll
    for (int kc = 0; kc < 16; kc++) {
      v16bf kb0 = frag_b(Kbuf[cur], 0,  512, kc * 32, lane);
      v16bf kb1 = frag_b(Kbuf[cur], 16, 512, kc * 32, lane);
      s0 = wmma_bf16(qf[kc], kb0, s0);
      s1 = wmma_bf16(qf[kc], kb1, s1);
    }

    const bool needMask = (t0 + 31) > q0;

    // ---- online softmax (rows i + 8*half live across the 16-lane half) ----
    #pragma unroll
    for (int i = 0; i < 8; i++) {
      const int srow = q0 + i + 8 * half;
      float v0 = s0[i] * sm, v1 = s1[i] * sm;
      if (needMask) {
        if (t0 + l16 > srow)      v0 = -1e30f;
        if (t0 + 16 + l16 > srow) v1 = -1e30f;
      }
      float rm = fmaxf(v0, v1);
      #pragma unroll
      for (int w = 1; w < 16; w <<= 1) rm = fmaxf(rm, __shfl_xor(rm, w, 32));
      const float mn = fmaxf(mrow[i], rm);
      const float alpha = __expf(mrow[i] - mn);
      mrow[i] = mn;
      const float p0 = __expf(v0 - mn);
      const float p1 = __expf(v1 - mn);
      float rs = p0 + p1;
      #pragma unroll
      for (int w = 1; w < 16; w <<= 1) rs += __shfl_xor(rs, w, 32);
      lrow[i] = lrow[i] * alpha + rs;
      #pragma unroll
      for (int n = 0; n < 8; n++) acc[n][i] *= alpha;

      const int prow = i + 8 * half;
      pT[prow * 32 + l16]      = (__bf16)p0;
      pT[prow * 32 + 16 + l16] = (__bf16)p1;
    }
    __syncthreads();

    // ---- reload P as an A-layout 16x32 fragment from LDS ----
    v16bf pf;
    {
      const __bf16* p = &pT[l16 * 32 + 8 * half];
      Frag32B f;
      f.u[0] = *(const uint4*)(p);
      f.u[1] = *(const uint4*)(p + 16);
      pf = f.v;
    }

    // ---- O += P @ V  (V tile LDS-resident, contiguous along t) ----
    #pragma unroll
    for (int n = 0; n < 8; n++) {
      v16bf vf = frag_b(Vbuf[cur], n * 16, 32, 0, lane);
      acc[n] = wmma_bf16(pf, vf, acc[n]);
    }

    async_wait0();  // next tile's async copies done; also fences pT reuse
  }

  // ---- normalize + store ctx tile (bf16) ----
  __bf16* C = ctx + ((size_t)b * S + q0) * D + h * DH;
  #pragma unroll
  for (int i = 0; i < 8; i++) {
    const float inv = 1.0f / lrow[i];
    const int row = i + 8 * half;
    #pragma unroll
    for (int n = 0; n < 8; n++)
      C[(size_t)row * D + n * 16 + l16] = (__bf16)(acc[n][i] * inv);
  }
}

// ---------------------------------------------------------------------------
// Host orchestration
// ---------------------------------------------------------------------------
extern "C" void kernel_launch(void* const* d_in, const int* in_sizes, int n_in,
                              void* d_out, int out_size, void* d_ws,
                              size_t ws_size, hipStream_t stream) {
  (void)in_sizes; (void)n_in; (void)out_size; (void)ws_size;

  const float* x     = (const float*)d_in[0];
  const float* Wq    = (const float*)d_in[1];
  const float* Wdkv  = (const float*)d_in[2];
  const float* Wuv   = (const float*)d_in[3];
  const float* Wuk   = (const float*)d_in[4];
  const float* Wo    = (const float*)d_in[5];
  const float* gamma = (const float*)d_in[6];
  const float* beta  = (const float*)d_in[7];

  const int Bb = 2, S = 2048, D = 2048, Hh = 16, DH = 128, L = 512;

  char* ws = (char*)d_ws;
  size_t off = 0;
  auto take = [&](size_t bytes) -> char* {
    char* p = ws + off;
    off += (bytes + 255) & ~(size_t)255;
    return p;
  };

  __bf16* x_bf    = (__bf16*)take((size_t)Bb * S * D * 2);
  __bf16* Wq_bf   = (__bf16*)take((size_t)D * D * 2);
  __bf16* Wdkv_bf = (__bf16*)take((size_t)L * D * 2);
  __bf16* Wuv_bf  = (__bf16*)take((size_t)D * L * 2);
  __bf16* WukT_bf = (__bf16*)take((size_t)L * D * 2);
  __bf16* Wo_bf   = (__bf16*)take((size_t)D * D * 2);
  float*  h_f32   = (float*)take((size_t)Bb * S * L * 4);
  __bf16* ckv_bf  = (__bf16*)take((size_t)Bb * S * L * 2);
  __bf16* akT_bf  = (__bf16*)take((size_t)L * D * 2);   // [L, D] = absorbed_k^T
  __bf16* tmp_bf  = (__bf16*)take((size_t)Bb * Hh * S * L * 2);
  __bf16* vT_bf   = (__bf16*)take((size_t)Bb * D * S * 2); // [b][d][t]
  __bf16* ctx_bf  = (__bf16*)take((size_t)Bb * S * D * 2);

  float* out_f   = (float*)d_out;                   // [B,S,D]
  float* ckv_out = out_f + (size_t)Bb * S * D;      // [B,S,L]

  // --- convert inputs to bf16 ---
  auto cvt = [&](const float* in, __bf16* o, size_t n) {
    f2bf_kernel<<<dim3((unsigned)((n + 255) / 256)), dim3(256), 0, stream>>>(in, o, n);
  };
  cvt(x,    x_bf,    (size_t)Bb * S * D);
  cvt(Wq,   Wq_bf,   (size_t)D * D);
  cvt(Wdkv, Wdkv_bf, (size_t)L * D);
  cvt(Wuv,  Wuv_bf,  (size_t)D * L);
  cvt(Wo,   Wo_bf,   (size_t)D * D);
  {
    size_t n = (size_t)L * D;
    trans_f2bf_kernel<<<dim3((unsigned)((n + 255) / 256)), dim3(256), 0, stream>>>(
        Wuk, WukT_bf, D, L);  // W_uk [D,L] -> [L,D]
  }

  // --- 1) h = x @ Wd_kv^T  -> [B*S, L] f32 ---
  gemm_bf16_wmma<<<dim3(L / 64, (Bb * S) / 64, 1), 32, 0, stream>>>(
      x_bf, Wdkv_bf, h_f32, D, D, D, L, 0, 0, 0, 0, 0, 0, 1, 0);

  // --- 2) c_kv = LN(h) -> f32 output tuple + bf16 ---
  ln_rows<<<dim3(Bb * S), 32, 0, stream>>>(h_f32, gamma, beta, ckv_out, ckv_bf);

  // --- 3) akT = (W_q @ W_uk)^T = W_uk^T @ W_q^T -> [L, D] bf16 ---
  gemm_bf16_wmma<<<dim3(D / 64, L / 64, 1), 32, 0, stream>>>(
      WukT_bf, Wq_bf, akT_bf, D, D, D, D, 0, 0, 0, 0, 0, 0, 1, 1);

  // --- 4) tmp[b,h] = q[b,:,h,:] @ ak[h]^T -> [B,H,S,L] bf16 ---
  gemm_bf16_wmma<<<dim3(L / 64, S / 64, Bb * Hh), 32, 0, stream>>>(
      x_bf, akT_bf, tmp_bf, DH, D, D, L,
      (long long)S * D, (long long)DH,         // A: x strided by (b,h)
      0, (long long)DH,                        // B: akT col-offset per head
      (long long)Hh * S * L, (long long)S * L, // C: tmp[b][h]
      Hh, 1);

  // --- 5) vT[b] = W_uv @ c_kv[b]^T -> [b][D][S] bf16 (V transposed) ---
  gemm_bf16_wmma<<<dim3(S / 64, D / 64, Bb), 32, 0, stream>>>(
      Wuv_bf, ckv_bf, vT_bf, L, L, L, S,
      0, 0, (long long)S * L, 0, (long long)D * S, 0, 1, 1);

  // --- 6) flash attention -> ctx [B,S,D] bf16 ---
  flash_mla<<<dim3(S / 16, Hh, Bb), 32, 0, stream>>>(tmp_bf, ckv_bf, vT_bf, ctx_bf);

  // --- 7) out = ctx @ W_o^T -> [B*S, D] f32 (into d_out) ---
  gemm_bf16_wmma<<<dim3(D / 64, (Bb * S) / 64, 1), 32, 0, stream>>>(
      ctx_bf, Wo_bf, out_f, D, D, D, D, 0, 0, 0, 0, 0, 0, 1, 0);
}
